// Topology_Extraction_53858889892102
// MI455X (gfx1250) — matrix operand order repaired and checked by reading, not compiled, not run on hardware
//
#include <hip/hip_runtime.h>
#include <hip/hip_bf16.h>
#include <stdint.h>

#define N_NODES 100000
#define N_EDGES 1600000
#define EPS 1e-5f

typedef __attribute__((ext_vector_type(16))) __bf16 bf16x16;
typedef __attribute__((ext_vector_type(8)))  float  f32x8;
typedef __attribute__((address_space(3))) unsigned short lds_ushort;

union AFrag { uint4 q[2]; bf16x16 v; };

__device__ __forceinline__ unsigned short f2bf(float f) {
    union { float f; uint32_t u; } c; c.f = f;
    uint32_t u = c.u;
    u += 0x7fffu + ((u >> 16) & 1u);   // round-to-nearest-even
    return (unsigned short)(u >> 16);
}

__device__ __forceinline__ void gAtomicAdd(float* p, float v) {
    unsafeAtomicAdd(p, v);             // hardware global_atomic_add_f32
}

// ---------- weight convert + transpose: src f32 [K][O] -> dst bf16 [O][K] ----------
__global__ void k_wcvt(const float* __restrict__ src, unsigned short* __restrict__ dst,
                       int K, int O) {
    int t = blockIdx.x * blockDim.x + threadIdx.x;
    if (t >= K * O) return;
    int k = t / O, o = t - k * O;
    dst[o * K + k] = f2bf(src[t]);
}

// ---------- plain f32 -> bf16 row copy (for async-staged MLP input) ----------
__global__ void k_f32_to_bf16(const float* __restrict__ src,
                              unsigned short* __restrict__ dst, int n4) {
    int t = blockIdx.x * blockDim.x + threadIdx.x;
    if (t >= n4) return;
    float4 v = *(const float4*)(src + (size_t)t * 4);
    *(ushort4*)(dst + (size_t)t * 4) =
        make_ushort4(f2bf(v.x), f2bf(v.y), f2bf(v.z), f2bf(v.w));
}

// ---------- in-degree count ----------
__global__ void k_degree(const int* __restrict__ di, float* __restrict__ deg) {
    int t = blockIdx.x * blockDim.x + threadIdx.x;
    if (t < N_EDGES) gAtomicAdd(&deg[di[t]], 1.0f);
}

// ---------- scatter-add of 64-channel features: agg[dst] += feat[src] ----------
__global__ void k_scatter64(const int* __restrict__ si, const int* __restrict__ di,
                            const float* __restrict__ feat, float* __restrict__ agg) {
    int t = blockIdx.x * blockDim.x + threadIdx.x;          // E * 16 threads
    if (t >= N_EDGES * 16) return;
    int e = t >> 4, q = (t & 15) << 2;
    int s = si[e], d = di[e];
    float4 v = *(const float4*)(feat + (size_t)s * 64 + q);
    float* p = agg + (size_t)d * 64 + q;
    gAtomicAdd(p + 0, v.x);
    gAtomicAdd(p + 1, v.y);
    gAtomicAdd(p + 2, v.z);
    gAtomicAdd(p + 3, v.w);
}

// ---------- fused SAGE linear: z = mean @ Wl + bl + x @ Wr  (bf16 WMMA) ----------
template<int OUT>
__global__ void k_sage(const float* __restrict__ xin, const float* __restrict__ agg,
                       const float* __restrict__ deg,
                       const unsigned short* __restrict__ Wlt,   // [OUT][64] bf16
                       const unsigned short* __restrict__ Wrt,   // [OUT][64] bf16
                       const float* __restrict__ bias,
                       float* __restrict__ z) {
    __shared__ unsigned short sM[32 * 64];
    __shared__ unsigned short sX[32 * 64];
    const int tid = threadIdx.x;
    const int n0  = blockIdx.x * 32;

    { // stage 32 rows of mean + x as bf16 into LDS (8 threads/row, 8 ch each)
        int row = tid >> 3;
        int cb  = (tid & 7) * 8;
        float dv  = deg[n0 + row];
        float inv = 1.0f / fmaxf(dv, 1.0f);
        const float* ap = agg + (size_t)(n0 + row) * 64 + cb;
        const float* xp = xin + (size_t)(n0 + row) * 64 + cb;
        float4 a0 = *(const float4*)(ap);
        float4 a1 = *(const float4*)(ap + 4);
        float4 x0 = *(const float4*)(xp);
        float4 x1 = *(const float4*)(xp + 4);
        *(ushort4*)(sM + row * 64 + cb) =
            make_ushort4(f2bf(a0.x * inv), f2bf(a0.y * inv), f2bf(a0.z * inv), f2bf(a0.w * inv));
        *(ushort4*)(sM + row * 64 + cb + 4) =
            make_ushort4(f2bf(a1.x * inv), f2bf(a1.y * inv), f2bf(a1.z * inv), f2bf(a1.w * inv));
        *(ushort4*)(sX + row * 64 + cb) =
            make_ushort4(f2bf(x0.x), f2bf(x0.y), f2bf(x0.z), f2bf(x0.w));
        *(ushort4*)(sX + row * 64 + cb + 4) =
            make_ushort4(f2bf(x1.x), f2bf(x1.y), f2bf(x1.z), f2bf(x1.w));
    }
    __syncthreads();

    const int wave = tid >> 5, lane = tid & 31;
    const int l16 = lane & 15, half = lane >> 4;
    const int ntiles = 2 * (OUT / 16);
    if (wave < ntiles) {            // wave-uniform branch: EXEC stays all ones
        int mt = wave & 1, nt = wave >> 1;
        int row = mt * 16 + l16;
        int col = nt * 16 + l16;
        f32x8 acc = {0.f, 0.f, 0.f, 0.f, 0.f, 0.f, 0.f, 0.f};
        #pragma unroll
        for (int ks = 0; ks < 2; ++ks) {
            int k0 = ks * 32;
            AFrag aM, aX, bL, bR;
            const unsigned short* mp = sM + row * 64 + k0 + half * 8;
            aM.q[0] = *(const uint4*)(mp);      aM.q[1] = *(const uint4*)(mp + 16);
            const unsigned short* xp = sX + row * 64 + k0 + half * 8;
            aX.q[0] = *(const uint4*)(xp);      aX.q[1] = *(const uint4*)(xp + 16);
            const unsigned short* lp = Wlt + col * 64 + k0 + half * 8;
            bL.q[0] = *(const uint4*)(lp);      bL.q[1] = *(const uint4*)(lp + 16);
            const unsigned short* rp = Wrt + col * 64 + k0 + half * 8;
            bR.q[0] = *(const uint4*)(rp);      bR.q[1] = *(const uint4*)(rp + 16);
            acc = __builtin_amdgcn_wmma_f32_16x16x32_bf16(false, aM.v, false, bL.v,
                                                          (short)0, acc, false, false);
            acc = __builtin_amdgcn_wmma_f32_16x16x32_bf16(false, aX.v, false, bR.v,
                                                          (short)0, acc, false, false);
        }
        float b = bias[col];
        #pragma unroll
        for (int v = 0; v < 8; ++v) {
            int r = mt * 16 + v + half * 8;
            z[(size_t)(n0 + r) * OUT + col] = acc[v] + b;
        }
    }
}

// ---------- BN statistics: stats[0..C)=sum, stats[C..2C)=sumsq ----------
template<int C>
__global__ void k_bn_stats(const float* __restrict__ z, float* __restrict__ stats) {
    __shared__ float sred[2 * C];
    const int tid = threadIdx.x;
    if (tid < 2 * C) sred[tid] = 0.0f;
    __syncthreads();
    const int G = C / 4;
    const int g = tid & (G - 1);
    const int r = tid / G;
    const int rowsPer = 256 / G;
    float4 s = make_float4(0.f, 0.f, 0.f, 0.f);
    float4 q = make_float4(0.f, 0.f, 0.f, 0.f);
    for (int n = blockIdx.x * rowsPer + r; n < N_NODES; n += gridDim.x * rowsPer) {
        float4 v = *(const float4*)(z + (size_t)n * C + g * 4);
        s.x += v.x; s.y += v.y; s.z += v.z; s.w += v.w;
        q.x += v.x * v.x; q.y += v.y * v.y; q.z += v.z * v.z; q.w += v.w * v.w;
    }
    atomicAdd(&sred[g * 4 + 0], s.x);  atomicAdd(&sred[g * 4 + 1], s.y);   // ds_add_f32
    atomicAdd(&sred[g * 4 + 2], s.z);  atomicAdd(&sred[g * 4 + 3], s.w);
    atomicAdd(&sred[C + g * 4 + 0], q.x);  atomicAdd(&sred[C + g * 4 + 1], q.y);
    atomicAdd(&sred[C + g * 4 + 2], q.z);  atomicAdd(&sred[C + g * 4 + 3], q.w);
    __syncthreads();
    if (tid < 2 * C) gAtomicAdd(&stats[tid], sred[tid]);
}

// ---------- apply BN + ReLU ----------
template<int C>
__global__ void k_bn_relu(const float* __restrict__ z, const float* __restrict__ stats,
                          const float* __restrict__ gamma, const float* __restrict__ beta,
                          float* __restrict__ out) {
    const float invN = 1.0f / (float)N_NODES;
    const int G = C / 4;
    int idx = blockIdx.x * blockDim.x + threadIdx.x;        // over N*C/4
    if (idx >= N_NODES * G) return;
    int g = idx % G;
    float4 v = *(const float4*)(z + (size_t)idx * 4);
    float  r[4] = {v.x, v.y, v.z, v.w};
    float  o[4];
    #pragma unroll
    for (int j = 0; j < 4; ++j) {
        int c = g * 4 + j;
        float mu  = stats[c] * invN;
        float var = stats[C + c] * invN - mu * mu;
        float sc  = rsqrtf(var + EPS) * gamma[c];
        o[j] = fmaxf((r[j] - mu) * sc + beta[c], 0.0f);
    }
    *(float4*)(out + (size_t)idx * 4) = make_float4(o[0], o[1], o[2], o[3]);
}

// ---------- fused MLP 32->1024->1024->16 + softmax, 32 rows per block ----------
// Input tile arrives as bf16 via per-lane async global->LDS DMA (ASYNCcnt path).
__global__ void k_mlp(const unsigned short* __restrict__ h2bf,
                      const unsigned short* __restrict__ W1t, const float* __restrict__ b1,
                      const unsigned short* __restrict__ W2t, const float* __restrict__ b2,
                      const unsigned short* __restrict__ W3t, const float* __restrict__ b3,
                      float* __restrict__ outSm, float* __restrict__ outLg) {
    extern __shared__ char smem[];
    unsigned short* sA = (unsigned short*)smem;                       // 32*32 bf16
    unsigned short* s1 = (unsigned short*)(smem + 2048);              // 32*1024 bf16
    unsigned short* s2 = (unsigned short*)(smem + 2048 + 65536);      // 32*1024 bf16
    float*        slog = (float*)(smem + 2048 + 2 * 65536);           // 32*16 f32

    const int tid = threadIdx.x;
    const int n0  = blockIdx.x * 32;

    { // async DMA: 8 bytes/lane, 2 KB tile total, global -> LDS, no VGPR round-trip
        lds_ushort* lp = (lds_ushort*)(sA + tid * 4);
        unsigned int ldsaddr = (unsigned int)(size_t)lp;
        unsigned long long ga =
            (unsigned long long)(h2bf + (size_t)n0 * 32 + (size_t)tid * 4);
        asm volatile("global_load_async_to_lds_b64 %0, %1, off"
                     :: "v"(ldsaddr), "v"(ga) : "memory");
        asm volatile("s_wait_asynccnt 0x0" ::: "memory");
    }
    __syncthreads();

    const int wave = tid >> 5, lane = tid & 31;
    const int l16 = lane & 15, half = lane >> 4;

    // ---- layer 1: [32x32] @ W1t[1024][32] -> s1, relu ----
    for (int t = wave; t < 128; t += 8) {
        int mt = t & 1, nt = t >> 1;
        int row = mt * 16 + l16;
        int col = nt * 16 + l16;
        AFrag fa, fb;
        const unsigned short* ap = sA + row * 32 + half * 8;
        fa.q[0] = *(const uint4*)(ap);     fa.q[1] = *(const uint4*)(ap + 16);
        const unsigned short* bp = W1t + col * 32 + half * 8;
        fb.q[0] = *(const uint4*)(bp);     fb.q[1] = *(const uint4*)(bp + 16);
        f32x8 acc = {0.f, 0.f, 0.f, 0.f, 0.f, 0.f, 0.f, 0.f};
        acc = __builtin_amdgcn_wmma_f32_16x16x32_bf16(false, fa.v, false, fb.v,
                                                      (short)0, acc, false, false);
        float bb = b1[col];
        #pragma unroll
        for (int v = 0; v < 8; ++v) {
            int r = mt * 16 + v + half * 8;
            s1[r * 1024 + col] = f2bf(fmaxf(acc[v] + bb, 0.0f));
        }
    }
    __syncthreads();

    // ---- layer 2: [32x1024] @ W2t[1024][1024] -> s2, relu (L2-resident weights) ----
    for (int t = wave; t < 128; t += 8) {
        int mt = t & 1, nt = t >> 1;
        int row = mt * 16 + l16;
        int col = nt * 16 + l16;
        const unsigned short* wrow = W2t + (size_t)col * 1024;
        f32x8 acc = {0.f, 0.f, 0.f, 0.f, 0.f, 0.f, 0.f, 0.f};
        for (int ks = 0; ks < 32; ++ks) {
            if (ks < 31) __builtin_prefetch(wrow + (ks + 1) * 32 + half * 16, 0, 0);
            AFrag fa, fb;
            const unsigned short* ap = s1 + row * 1024 + ks * 32 + half * 8;
            fa.q[0] = *(const uint4*)(ap);  fa.q[1] = *(const uint4*)(ap + 16);
            const unsigned short* bp = wrow + ks * 32 + half * 8;
            fb.q[0] = *(const uint4*)(bp);  fb.q[1] = *(const uint4*)(bp + 16);
            acc = __builtin_amdgcn_wmma_f32_16x16x32_bf16(false, fa.v, false, fb.v,
                                                          (short)0, acc, false, false);
        }
        float bb = b2[col];
        #pragma unroll
        for (int v = 0; v < 8; ++v) {
            int r = mt * 16 + v + half * 8;
            s2[r * 1024 + col] = f2bf(fmaxf(acc[v] + bb, 0.0f));
        }
    }
    __syncthreads();

    // ---- layer 3: [32x1024] @ W3t[16][1024] -> logits in LDS ----
    if (wave < 2) {
        int mt  = wave;
        int row = mt * 16 + l16;
        int col = l16;
        const unsigned short* wrow = W3t + (size_t)col * 1024;
        f32x8 acc = {0.f, 0.f, 0.f, 0.f, 0.f, 0.f, 0.f, 0.f};
        for (int ks = 0; ks < 32; ++ks) {
            AFrag fa, fb;
            const unsigned short* ap = s2 + row * 1024 + ks * 32 + half * 8;
            fa.q[0] = *(const uint4*)(ap);  fa.q[1] = *(const uint4*)(ap + 16);
            const unsigned short* bp = wrow + ks * 32 + half * 8;
            fb.q[0] = *(const uint4*)(bp);  fb.q[1] = *(const uint4*)(bp + 16);
            acc = __builtin_amdgcn_wmma_f32_16x16x32_bf16(false, fa.v, false, fb.v,
                                                          (short)0, acc, false, false);
        }
        float bb = b3[col];
        #pragma unroll
        for (int v = 0; v < 8; ++v) {
            int r = mt * 16 + v + half * 8;
            slog[r * 16 + col] = acc[v] + bb;
        }
    }
    __syncthreads();

    // ---- softmax + write logits/softmax ----
    if (tid < 32) {
        float lv[16];
        float mx = -1e30f;
        #pragma unroll
        for (int j = 0; j < 16; ++j) { lv[j] = slog[tid * 16 + j]; mx = fmaxf(mx, lv[j]); }
        float sum = 0.0f;
        #pragma unroll
        for (int j = 0; j < 16; ++j) { lv[j] = __expf(lv[j] - mx); sum += lv[j]; }
        float inv = 1.0f / sum;
        float* lg = outLg + (size_t)(n0 + tid) * 16;
        float* sm = outSm + (size_t)(n0 + tid) * 16;
        #pragma unroll
        for (int j = 0; j < 16; ++j) {
            lg[j] = slog[tid * 16 + j];
            sm[j] = lv[j] * inv;
        }
    }
}

extern "C" void kernel_launch(void* const* d_in, const int* in_sizes, int n_in,
                              void* d_out, int out_size, void* d_ws, size_t ws_size,
                              hipStream_t stream) {
    const float* x   = (const float*)d_in[0];
    const int*   ei  = (const int*)d_in[1];
    const float* W1l = (const float*)d_in[2];
    const float* b1l = (const float*)d_in[3];
    const float* W1r = (const float*)d_in[4];
    const float* g1  = (const float*)d_in[5];
    const float* be1 = (const float*)d_in[6];
    const float* W2l = (const float*)d_in[7];
    const float* b2l = (const float*)d_in[8];
    const float* W2r = (const float*)d_in[9];
    const float* g2  = (const float*)d_in[10];
    const float* be2 = (const float*)d_in[11];
    const float* Wm1 = (const float*)d_in[12];
    const float* bm1 = (const float*)d_in[13];
    const float* Wm2 = (const float*)d_in[14];
    const float* bm2 = (const float*)d_in[15];
    const float* Wm3 = (const float*)d_in[16];
    const float* bm3 = (const float*)d_in[17];
    (void)in_sizes; (void)n_in; (void)out_size; (void)ws_size;

    const int* src = ei;
    const int* dst = ei + N_EDGES;

    float* outSm = (float*)d_out;
    float* outLg = outSm + (size_t)N_NODES * 16;
    float* h1    = outSm + (size_t)N_NODES * 32;   // x_temp_1 [N,64]
    float* h2    = outSm + (size_t)N_NODES * 96;   // x_temp_2 [N,32]

    // workspace carve-out (256B aligned)
    char* ws = (char*)d_ws;
    size_t off = 0;
    auto take = [&](size_t bytes) -> char* {
        char* p = ws + off;
        off = (off + bytes + 255) & ~(size_t)255;
        return p;
    };
    float* deg   = (float*)take((size_t)N_NODES * 4);
    float* agg   = (float*)take((size_t)N_NODES * 64 * 4);
    float* z     = (float*)take((size_t)N_NODES * 64 * 4);
    float* stats = (float*)take(2 * 64 * 4);
    unsigned short* W1lt = (unsigned short*)take(64 * 64 * 2);
    unsigned short* W1rt = (unsigned short*)take(64 * 64 * 2);
    unsigned short* W2lt = (unsigned short*)take(32 * 64 * 2);
    unsigned short* W2rt = (unsigned short*)take(32 * 64 * 2);
    unsigned short* Wm1t = (unsigned short*)take(1024 * 32 * 2);
    unsigned short* Wm2t = (unsigned short*)take((size_t)1024 * 1024 * 2);
    unsigned short* Wm3t = (unsigned short*)take(16 * 1024 * 2);
    unsigned short* h2bf = (unsigned short*)take((size_t)N_NODES * 32 * 2);

    auto cvt = [&](const float* s, unsigned short* d, int K, int O) {
        int tot = K * O;
        k_wcvt<<<(tot + 255) / 256, 256, 0, stream>>>(s, d, K, O);
    };
    cvt(W1l, W1lt, 64, 64);
    cvt(W1r, W1rt, 64, 64);
    cvt(W2l, W2lt, 64, 32);
    cvt(W2r, W2rt, 64, 32);
    cvt(Wm1, Wm1t, 32, 1024);
    cvt(Wm2, Wm2t, 1024, 1024);
    cvt(Wm3, Wm3t, 1024, 16);

    hipMemsetAsync(deg, 0, (size_t)N_NODES * 4, stream);
    k_degree<<<(N_EDGES + 255) / 256, 256, 0, stream>>>(dst, deg);

    // ---- SAGE layer 1 ----
    hipMemsetAsync(agg, 0, (size_t)N_NODES * 64 * 4, stream);
    hipMemsetAsync(stats, 0, 2 * 64 * 4, stream);
    k_scatter64<<<(N_EDGES * 16) / 256, 256, 0, stream>>>(src, dst, x, agg);
    k_sage<64><<<N_NODES / 32, 256, 0, stream>>>(x, agg, deg, W1lt, W1rt, b1l, z);
    k_bn_stats<64><<<320, 256, 0, stream>>>(z, stats);
    k_bn_relu<64><<<(N_NODES * 16 + 255) / 256, 256, 0, stream>>>(z, stats, g1, be1, h1);

    // ---- SAGE layer 2 ----
    hipMemsetAsync(agg, 0, (size_t)N_NODES * 64 * 4, stream);
    hipMemsetAsync(stats, 0, 2 * 64 * 4, stream);
    k_scatter64<<<(N_EDGES * 16) / 256, 256, 0, stream>>>(src, dst, h1, agg);
    k_sage<32><<<N_NODES / 32, 256, 0, stream>>>(h1, agg, deg, W2lt, W2rt, b2l, z);
    k_bn_stats<32><<<320, 256, 0, stream>>>(z, stats);
    k_bn_relu<32><<<(N_NODES * 8 + 255) / 256, 256, 0, stream>>>(z, stats, g2, be2, h2);

    // bf16 sidecar of h2 for async LDS staging in the MLP
    k_f32_to_bf16<<<(N_NODES * 8 + 255) / 256, 256, 0, stream>>>(h2, h2bf, N_NODES * 8);

    // ---- fused MLP + softmax ----
    const int smemBytes = 2048 + 65536 + 65536 + 2048;   // 132 KB dynamic LDS
    k_mlp<<<N_NODES / 32, 256, smemBytes, stream>>>(h2bf, Wm1t, bm1, Wm2t, bm2,
                                                    Wm3t, bm3, outSm, outLg);
}
